// SingleMatchingGroup_6811818132507
// MI455X (gfx1250) — compile-verified, matrix operand order checked by reading, not run on hardware
//
#include <hip/hip_runtime.h>

// ---------------- problem constants (from reference) ----------------
#define BATCH   2
#define HH      64
#define WW      64
#define CIN     128          // IN_FIELDS * FS
#define COUT    256          // NC * FS
#define NC      32
#define FS      8
#define PATCH   5
#define KCONV1  1152         // 9 * CIN
#define NPIX    (BATCH*HH*WW)   // 8192

typedef unsigned int       u32;
typedef unsigned long long u64;
typedef __attribute__((ext_vector_type(16))) __bf16 v16bf;
typedef __attribute__((ext_vector_type(8)))  float  v8f;
typedef __attribute__((ext_vector_type(4)))  u32    su4;
typedef __attribute__((ext_vector_type(8)))  u32    su8;

union FragBF { v16bf v; uint4 q[2]; };

// ---------------- workspace layout (bytes) ----------------
// B1p: [36 kblocks][128 cols][32 k]  (fragment-native, K fastest)
// B2p: [4  kblocks][256 cols][32 k]
#define OFF_XBF   0
#define SZ_XBF    (NPIX*CIN*2)                 // 2 MB
#define OFF_B1    (OFF_XBF + SZ_XBF)
#define SZ_B1     (KCONV1*CIN*2)               // 288 KB
#define OFF_BIAS  (OFF_B1 + SZ_B1)
#define SZ_BIAS   (CIN*4)
#define OFF_H     (OFF_BIAS + SZ_BIAS)
#define SZ_H      (NPIX*CIN*2)                 // 2 MB
#define OFF_B2    (OFF_H + SZ_H)
#define SZ_B2     (CIN*COUT*2)                 // 64 KB
#define OFF_F     (OFF_B2 + SZ_B2)
#define SZ_F      (NPIX*COUT*2)                // 4 MB

// ---------------- TDM: 1-D contiguous global -> LDS DMA (bf16 elements) ----------
// D# per CDNA5 ISA. Tracked by TENSORcnt (async issue,
// explicit s_wait_tensorcnt at the consume point -> DMA overlaps WMMA work).
__device__ __forceinline__ void tdm_load_1d(u32 lds_addr, const void* gptr, u32 nelem) {
    u64 ga = (u64)gptr;
    su4 g0;
    g0.x = 1u;                                            // count=1 (user, load)
    g0.y = lds_addr;                                      // LDS byte address
    g0.z = (u32)ga;                                       // global_addr[31:0]
    g0.w = ((u32)(ga >> 32) & 0x01FFFFFFu) | (2u << 30);  // addr[56:32] | type=2
    su8 g1;
    g1.s0 = 0x00010000u;                   // wg_mask=0, data_size=1 (2 bytes)
    g1.s1 = (nelem & 0xFFFFu) << 16;       // tensor_dim0[15:0]   (abar_addr=0)
    g1.s2 = (nelem >> 16) | (1u << 16);    // tensor_dim0[31:16] | tensor_dim1=1
    g1.s3 = (nelem & 0xFFFFu) << 16;       // tensor_dim1[31:16]=0 | tile_dim0
    g1.s4 = 1u;                            // tile_dim1=1, tile_dim2=0
    g1.s5 = nelem;                         // tensor_dim0_stride[31:0]
    g1.s6 = 0u;                            // stride0[47:32]=0, stride1 lo=0
    g1.s7 = 0u;
    asm volatile("tensor_load_to_lds %0, %1" :: "s"(g0), "s"(g1) : "memory");
}

// ---------------- kernel 1: NCHW f32 -> NHWC bf16 ----------------
__global__ void k_convert_x(const float* __restrict__ x, __bf16* __restrict__ xbf) {
    int idx = blockIdx.x * blockDim.x + threadIdx.x;
    if (idx >= NPIX * CIN) return;
    int xp = idx & 63;
    int yv = (idx >> 6) & 63;
    int c  = (idx >> 12) & 127;
    int b  = idx >> 19;
    xbf[((b * HH + yv) * WW + xp) * CIN + c] = (__bf16)x[idx];
}

// ---------------- kernel 2: weight repack (fragment-native) + BN fold ----------------
__global__ void k_prep_w(const float* __restrict__ w1, const float* __restrict__ gamma,
                         const float* __restrict__ beta, const float* __restrict__ mean,
                         const float* __restrict__ var, const float* __restrict__ w2,
                         __bf16* __restrict__ B1p, float* __restrict__ bias,
                         __bf16* __restrict__ B2p) {
    int idx = blockIdx.x * blockDim.x + threadIdx.x;
    const int N1 = KCONV1 * CIN;          // 147456
    const int N2 = CIN * COUT;            // 32768
    if (idx < N1) {
        // dest: ((kb32*128 + n)*32 + kk)
        int kk   = idx & 31;
        int n    = (idx >> 5) & 127;
        int kb32 = idx >> 12;
        int k    = kb32 * 32 + kk;        // (dy*3+dx)*128 + ci
        int ci   = k & 127;
        int tap  = k >> 7;
        int dy = tap / 3, dx = tap % 3;
        float s = gamma[n >> 3] * rsqrtf(var[n >> 3] + 1e-5f);
        B1p[idx] = (__bf16)(w1[((n * CIN + ci) * 3 + dy) * 3 + dx] * s);
    } else if (idx < N1 + N2) {
        int j    = idx - N1;
        int kk   = j & 31;
        int n    = (j >> 5) & 255;        // cout
        int kb32 = j >> 13;
        int ci   = kb32 * 32 + kk;
        B2p[j] = (__bf16)(w2[n * CIN + ci]);
    } else if (idx < N1 + N2 + CIN) {
        int n = idx - N1 - N2;
        float s = gamma[n >> 3] * rsqrtf(var[n >> 3] + 1e-5f);
        bias[n] = beta[n >> 3] - mean[n >> 3] * s;
    }
}

// fragment load: row of 32 bf16 (64B = quads q0..q3); lane<16 -> q0,q2 ; lane>=16 -> q1,q3
__device__ __forceinline__ v16bf load_frag(const __bf16* rowbase, int qi) {
    FragBF f;
    const uint4* p = (const uint4*)rowbase;
    f.q[0] = p[qi];
    f.q[1] = p[2 + qi];
    return f.v;
}

// ---------------- kernel 3: conv1 implicit GEMM + BN + ReLU ----------------
// grid: 128 workgroups (one per (b,y) image row), 128 threads (4 waves x 16 rows)
// Software-pipelined: A staged by waves, B staged by TDM, both double-buffered.
__global__ __launch_bounds__(128) void k_conv1(const __bf16* __restrict__ xbf,
                                               const __bf16* __restrict__ B1p,
                                               const float*  __restrict__ bias,
                                               __bf16* __restrict__ h) {
    __shared__ __align__(16) __bf16 sA[2][64 * 32];     //  2 x 4 KB
    __shared__ __align__(16) __bf16 sB[2][128 * 32];    //  2 x 8 KB
    const int by = blockIdx.x;                // b*64 + y
    const int b  = by >> 6;
    const int y  = by & 63;
    const int tid  = threadIdx.x;
    const int wave = tid >> 5;
    const int lane = tid & 31;
    const int qi   = lane >> 4;               // which K-half this lane owns
    const int l15  = lane & 15;

    v8f acc[8];
    #pragma unroll
    for (int nt = 0; nt < 8; ++nt) acc[nt] = (v8f){0,0,0,0,0,0,0,0};

    const int m    = tid >> 1;                // pixel row this thread stages
    const int half = (tid & 1) * 16;

    // stage A tile for kblk into buffer buf (im2row gather with zero pad)
    auto stageA = [&](int kblk, int buf) {
        int tap = kblk >> 2;
        int c0  = (kblk & 3) << 5;
        int dy  = tap / 3;
        int dx  = tap - dy * 3;
        int yy  = y + dy - 1;
        int xx  = m + dx - 1;
        uint4 d0 = {0,0,0,0}, d1 = {0,0,0,0};
        if (yy >= 0 && yy < HH && xx >= 0 && xx < WW) {
            const uint4* s4 = (const uint4*)(xbf + (((b * HH + yy) * WW + xx) * CIN + c0 + half));
            d0 = s4[0]; d1 = s4[1];
        }
        uint4* a4 = (uint4*)&sA[buf][m * 32 + half];
        a4[0] = d0; a4[1] = d1;
    };

    // prologue: fill buffer 0
    stageA(0, 0);
    if (wave == 0)
        tdm_load_1d((u32)(u64)&sB[0][0], B1p, 128 * 32);

    for (int i = 0; i < 36; ++i) {
        const int cur = i & 1, nxt = cur ^ 1;
        if (wave == 0)
            __builtin_amdgcn_s_wait_tensorcnt(0);     // sB[cur] DMA complete
        __syncthreads();                              // publish sA[cur], sB[cur]
        // issue next tiles: DMA + A staging overlap this iteration's WMMAs
        if (i + 1 < 36) {
            stageA(i + 1, nxt);
            if (wave == 0)
                tdm_load_1d((u32)(u64)&sB[nxt][0], B1p + (i + 1) * (128 * 32), 128 * 32);
        }
        const int arow = wave * 16 + l15;
        v16bf av = load_frag(&sA[cur][arow * 32], qi);
        #pragma unroll
        for (int nt = 0; nt < 8; ++nt) {
            int col = nt * 16 + l15;
            v16bf bv = load_frag(&sB[cur][col * 32], qi);
            acc[nt] = __builtin_amdgcn_wmma_f32_16x16x32_bf16(
                false, av, false, bv, (short)0, acc[nt], false, false);
        }
    }

    // ---- epilogue: bias + ReLU + bf16 store (C layout: vgpr r <-> M=r / M=8+r)
    #pragma unroll
    for (int nt = 0; nt < 8; ++nt) {
        int col = nt * 16 + l15;
        float bsv = bias[col];
        #pragma unroll
        for (int r = 0; r < 8; ++r) {
            int row = wave * 16 + ((lane < 16) ? r : r + 8);
            float vv = acc[nt][r] + bsv;
            vv = vv > 0.0f ? vv : 0.0f;
            h[(by * 64 + row) * CIN + col] = (__bf16)vv;
        }
    }
}

// ---------------- kernel 4: conv2 (1x1) GEMM 8192x256x128 ----------------
__global__ __launch_bounds__(128) void k_conv2(const __bf16* __restrict__ h,
                                               const __bf16* __restrict__ B2p,
                                               __bf16* __restrict__ f) {
    __shared__ __align__(16) __bf16 sA[64 * 128];       // whole A tile   16 KB
    __shared__ __align__(16) __bf16 sB[2][256 * 32];    //  2 x 16 KB
    const int mb   = blockIdx.x * 64;
    const int tid  = threadIdx.x;
    const int wave = tid >> 5;
    const int lane = tid & 31;
    const int qi   = lane >> 4;
    const int l15  = lane & 15;

    v8f acc[16];
    #pragma unroll
    for (int nt = 0; nt < 16; ++nt) acc[nt] = (v8f){0,0,0,0,0,0,0,0};

    // prologue: A tile (contiguous 16 KB = 64 consecutive rows of h) + first B slice
    if (wave == 0) {
        tdm_load_1d((u32)(u64)&sA[0], h + mb * CIN, 64 * CIN);
        tdm_load_1d((u32)(u64)&sB[0][0], B2p, 256 * 32);
    }

    for (int kb = 0; kb < 4; ++kb) {
        const int cur = kb & 1, nxt = cur ^ 1;
        if (wave == 0)
            __builtin_amdgcn_s_wait_tensorcnt(0);     // A (first iter) + sB[cur] done
        __syncthreads();
        if (wave == 0 && kb + 1 < 4)
            tdm_load_1d((u32)(u64)&sB[nxt][0], B2p + (kb + 1) * (256 * 32), 256 * 32);

        const int arow = wave * 16 + l15;
        v16bf av = load_frag(&sA[arow * CIN + kb * 32], qi);
        #pragma unroll
        for (int nt = 0; nt < 16; ++nt) {
            int col = nt * 16 + l15;
            v16bf bv = load_frag(&sB[cur][col * 32], qi);
            acc[nt] = __builtin_amdgcn_wmma_f32_16x16x32_bf16(
                false, av, false, bv, (short)0, acc[nt], false, false);
        }
    }

    #pragma unroll
    for (int nt = 0; nt < 16; ++nt) {
        int col = nt * 16 + l15;
        #pragma unroll
        for (int r = 0; r < 8; ++r) {
            int row = wave * 16 + ((lane < 16) ? r : r + 8);
            f[(mb + row) * COUT + col] = (__bf16)acc[nt][r];
        }
    }
}

// ---------------- kernel 5: patch reflection similarity ----------------
#define VSTRIDE 201   // 200 + 1 pad -> lane LDS stride co-prime with 64 banks

__device__ __forceinline__ float sampF(const __bf16* f, int b, int yy, int xx, int ch) {
    if (yy < 0 || yy >= HH || xx < 0 || xx >= WW) return 0.0f;
    return (float)f[((b * HH + yy) * WW + xx) * COUT + ch];
}

// grid: 8192 blocks (one pixel), 64 threads (wave 0 = rot 0deg, wave 1 = rot -45deg)
__global__ __launch_bounds__(64) void k_sim(const __bf16* __restrict__ f,
                                            float* __restrict__ out) {
    __shared__ float vals[2 * 32 * VSTRIDE];            // ~50.3 KB
    const int pix  = blockIdx.x;
    const int b = pix >> 12;
    const int y = (pix >> 6) & 63;
    const int x = pix & 63;
    const int tid  = threadIdx.x;
    const int rot  = tid >> 5;          // rotation index 0/1
    const int lane = tid & 31;          // nc channel
    float* myv = &vals[(rot * 32 + lane) * VSTRIDE];

    const float ct =  0.70710678118654752f;   // cos(-45deg)
    const float st = -0.70710678118654752f;   // sin(-45deg)

    for (int fs = 0; fs < FS; ++fs) {
        int ch = lane * FS + fs;
        if (rot == 1) ch = (ch & ~3) | ((ch + 1) & 3);   // rot_perms[1]
        #pragma unroll
        for (int py = 0; py < PATCH; ++py) {
            #pragma unroll
            for (int px = 0; px < PATCH; ++px) {
                float val;
                if (rot == 0) {
                    val = sampF(f, b, y + py - 2, x + px - 2, ch);
                } else {
                    // static grid: everything below constant-folds
                    float dx = (float)px - 2.0f, dyv = (float)py - 2.0f;
                    float sx = ct * dx + st * dyv + 2.0f;
                    float sy = -st * dx + ct * dyv + 2.0f;
                    int x0 = (int)floorf(sx), y0 = (int)floorf(sy);
                    float wx = sx - (float)x0, wy = sy - (float)y0;
                    const float cw[4] = {(1.f - wy) * (1.f - wx), (1.f - wy) * wx,
                                         wy * (1.f - wx), wy * wx};
                    val = 0.0f;
                    #pragma unroll
                    for (int c4 = 0; c4 < 4; ++c4) {
                        int yi = y0 + (c4 >> 1);
                        int xi = x0 + (c4 & 1);
                        if (yi >= 0 && yi < PATCH && xi >= 0 && xi < PATCH && cw[c4] != 0.0f)
                            val += cw[c4] * sampF(f, b, y + yi - 2, x + xi - 2, ch);
                    }
                }
                myv[fs * 25 + py * 5 + px] = val;
            }
        }
    }
    // per-lane region only -> no cross-thread sharing, no barrier needed
    const int rp[FS] = {4, 7, 6, 5, 0, 3, 2, 1};          // refl_perm within FS group
    float dot = 0.0f, nsq = 0.0f;
    #pragma unroll
    for (int fs = 0; fs < FS; ++fs) {
        for (int py = 0; py < PATCH; ++py) {
            #pragma unroll
            for (int px = 0; px < PATCH; ++px) {
                float v  = myv[fs * 25 + py * 5 + px];
                float vr = myv[rp[fs] * 25 + py * 5 + (4 - px)];
                dot += v * vr;
                nsq += v * v;
            }
        }
    }
    float nrm = fmaxf(sqrtf(nsq), 1e-6f);
    float sim = dot / (nrm * nrm);
    // out channel c = nc*4 + r, value sim_{r%2}: write r = rot and rot+2
    out[((b * (NC * 4) + lane * 4 + rot) * HH + y) * WW + x]     = sim;
    out[((b * (NC * 4) + lane * 4 + rot + 2) * HH + y) * WW + x] = sim;
}

// ---------------- launcher ----------------
extern "C" void kernel_launch(void* const* d_in, const int* in_sizes, int n_in,
                              void* d_out, int out_size, void* d_ws, size_t ws_size,
                              hipStream_t stream) {
    const float* x     = (const float*)d_in[0];
    const float* w1    = (const float*)d_in[1];
    const float* gamma = (const float*)d_in[2];
    const float* beta  = (const float*)d_in[3];
    const float* mean  = (const float*)d_in[4];
    const float* var   = (const float*)d_in[5];
    const float* w2    = (const float*)d_in[6];

    unsigned char* ws = (unsigned char*)d_ws;
    __bf16* xbf  = (__bf16*)(ws + OFF_XBF);
    __bf16* B1p  = (__bf16*)(ws + OFF_B1);
    float*  bias = (float*)(ws + OFF_BIAS);
    __bf16* h    = (__bf16*)(ws + OFF_H);
    __bf16* B2p  = (__bf16*)(ws + OFF_B2);
    __bf16* f    = (__bf16*)(ws + OFF_F);

    k_convert_x<<<(NPIX * CIN + 255) / 256, 256, 0, stream>>>(x, xbf);
    k_prep_w<<<(KCONV1 * CIN + CIN * COUT + CIN + 255) / 256, 256, 0, stream>>>(
        w1, gamma, beta, mean, var, w2, B1p, bias, B2p);
    k_conv1<<<BATCH * HH, 128, 0, stream>>>(xbf, B1p, bias, h);
    k_conv2<<<NPIX / 64, 128, 0, stream>>>(h, B2p, f);
    k_sim<<<NPIX, 64, 0, stream>>>(f, (float*)d_out);
}
